// CustomEncoder_377957122357
// MI455X (gfx1250) — matrix operand order repaired and checked by reading, not compiled, not run on hardware
//
#include <hip/hip_runtime.h>

// ---------------- types ----------------
typedef __bf16 bf16_t;
typedef __attribute__((ext_vector_type(16))) __bf16 v16bf;
typedef __attribute__((ext_vector_type(8)))  __bf16 v8bf;
typedef __attribute__((ext_vector_type(8)))  float  v8f;

// ---------------- problem constants ----------------
#define Sn  64
#define Bn  128
#define Vn  32000
#define En  500
#define Hn  500
#define Ep  512          // E padded for K-dim
#define Hp  512          // H padded for K-dim
#define H2p 1024         // 2H padded
#define Gn  2000         // 4H gate width
#define Gp  2048         // gate width padded (also padded N rows for weights)

__device__ inline float sigm(float x) { return 1.0f / (1.0f + __expf(-x)); }

// ---------------- WMMA fragment loaders (wave32 layouts, ISA 7.12.2) ----------------
// A 16x32 bf16: lane holds row (lane&15); K runs {k0+8*(lane>>4)+[0..7], +16}
__device__ inline v16bf load_afrag(const bf16_t* __restrict__ p) {
  v8bf lo = *(const v8bf*)p;
  v8bf hi = *(const v8bf*)(p + 16);
  v16bf a;
#pragma unroll
  for (int i = 0; i < 8; ++i) { a[i] = lo[i]; a[i + 8] = hi[i]; }
  return a;
}
// B 32x16 bf16 stored as (N,K): lane holds col (lane&15); K runs {k0+8*(lane>>4)+[0..7], +16}
__device__ inline v16bf load_bfrag(const bf16_t* __restrict__ p) {
  v8bf lo = *(const v8bf*)p;
  v8bf hi = *(const v8bf*)(p + 16);
  v16bf b;
#pragma unroll
  for (int i = 0; i < 8; ++i) { b[i] = lo[i]; b[i + 8] = hi[i]; }
  return b;
}

#define WMMA_BF16(A, B, C) \
  __builtin_amdgcn_wmma_f32_16x16x32_bf16(false, (A), false, (B), (short)0, (C), false, false)

// ---- register-blocked accumulation: wave owns 2 M-tiles x 4 N-tiles (32x64) ----
// Compile-time LDA/LDB/K so B-row strides and K-steps fold into load immediates
// (no per-iteration 64-bit address adds, no VGPR spill).
template <int LDA, int LDB, int KK>
__device__ inline void wmma_2x4(const bf16_t* __restrict__ A,
                                const bf16_t* __restrict__ Bw,
                                int rowBase, int colBase, v8f acc[8]) {
  const int lane = threadIdx.x & 31;
  const int half = lane >> 4;
  const int l15  = lane & 15;
  const bf16_t* ap = A + (size_t)(rowBase + l15) * LDA + half * 8;
  const bf16_t* bp = Bw + (size_t)(colBase + l15) * LDB + half * 8;
#pragma unroll 4
  for (int k0 = 0; k0 < KK; k0 += 32) {
    v16bf a0 = load_afrag(ap + k0);
    v16bf a1 = load_afrag(ap + 16 * LDA + k0);
    v16bf b0 = load_bfrag(bp + k0);
    v16bf b1 = load_bfrag(bp + 16 * LDB + k0);
    v16bf b2 = load_bfrag(bp + 32 * LDB + k0);
    v16bf b3 = load_bfrag(bp + 48 * LDB + k0);
    acc[0] = WMMA_BF16(a0, b0, acc[0]);
    acc[1] = WMMA_BF16(a0, b1, acc[1]);
    acc[2] = WMMA_BF16(a0, b2, acc[2]);
    acc[3] = WMMA_BF16(a0, b3, acc[3]);
    acc[4] = WMMA_BF16(a1, b0, acc[4]);
    acc[5] = WMMA_BF16(a1, b1, acc[5]);
    acc[6] = WMMA_BF16(a1, b2, acc[6]);
    acc[7] = WMMA_BF16(a1, b3, acc[7]);
  }
}

__device__ inline void store_f32_2x4(float* __restrict__ C, int ldc,
                                     int rowBase, int colBase,
                                     const float* __restrict__ bias, int nBias,
                                     int nTilesValid, const v8f acc[8]) {
  const int lane = threadIdx.x & 31;
  const int half = lane >> 4;
  const int l15  = lane & 15;
#pragma unroll
  for (int mt = 0; mt < 2; ++mt)
#pragma unroll
    for (int nt = 0; nt < 4; ++nt) {
      int ntile = colBase / 16 + nt;
      if (ntile >= nTilesValid) continue;         // wave-uniform
      int n  = colBase + nt * 16 + l15;
      float bv = (bias != nullptr && n < nBias) ? bias[n] : 0.0f;
      int m0 = rowBase + mt * 16 + half * 8;
      v8f a = acc[mt * 4 + nt];
#pragma unroll
      for (int r = 0; r < 8; ++r)
        C[(size_t)(m0 + r) * ldc + n] = a[r] + bv;
    }
}

// ---------------- utility kernels ----------------
__global__ void k_zero_u32(unsigned int* p, size_t n) {
  size_t i = (size_t)blockIdx.x * blockDim.x + threadIdx.x;
  size_t stride = (size_t)gridDim.x * blockDim.x;
  for (; i < n; i += stride) p[i] = 0u;
}

// pack fp32 weight into (Npad, Kpad) row-major bf16, zero padded.
// transpose=0: src is (N,K) row-major.  transpose=1: src is (K,N) row-major.
__global__ void k_pack(const float* __restrict__ src, bf16_t* __restrict__ dst,
                       int N, int K, int Npad, int Kpad, int transpose) {
  size_t idx = (size_t)blockIdx.x * blockDim.x + threadIdx.x;
  size_t tot = (size_t)Npad * Kpad;
  if (idx >= tot) return;
  int n = (int)(idx / Kpad), k = (int)(idx % Kpad);
  float v = 0.0f;
  if (n < N && k < K)
    v = transpose ? src[(size_t)k * N + n] : src[(size_t)n * K + k];
  dst[idx] = (bf16_t)v;
}

__global__ void k_bias_add(const float* a, const float* b, float* dst, int n) {
  int i = blockIdx.x * blockDim.x + threadIdx.x;
  if (i < n) dst[i] = a[i] + b[i];
}

// gather: Eg[(s,k,b), e<512] = bf16(emb[k, tok(s,b), e]) (0 for e>=E)
__global__ void k_gather(const int* __restrict__ tok, const float* __restrict__ emb,
                         bf16_t* __restrict__ Eg) {
  int blk = blockIdx.x;                    // (s*5 + k)*Bn + b
  int b = blk % Bn;
  int k = (blk / Bn) % 5;
  int s = blk / (Bn * 5);
  int t = tok[s * Bn + b];
  const float* src = emb + ((size_t)k * Vn + t) * En;
  bf16_t* dst = Eg + (size_t)blk * Ep;
  for (int e = threadIdx.x; e < Ep; e += blockDim.x)
    dst[e] = (e < En) ? (bf16_t)src[e] : (bf16_t)0.0f;
}

// ---------------- generic WMMA GEMM kernels ----------------
// block = 128 thr = 4 waves; wave: 2 Mtiles x 4 Ntiles.  grid (Mtiles/2, ceil(Nt/16), Z)
template <int LDA, int LDB, int KK>
__global__ void k_gemm_f32(const bf16_t* __restrict__ A, unsigned long long aSz,
                           const bf16_t* __restrict__ Bw, unsigned long long bSz,
                           const float* __restrict__ bias, int nBias,
                           float* __restrict__ C, int ldc, unsigned long long cSz,
                           int nTilesValid) {
  int wv = threadIdx.x >> 5;
  int ngroup = blockIdx.y * 4 + wv;                  // 4 N-tiles per group
  if (ngroup * 4 >= nTilesValid) return;             // wave-uniform
  int z = blockIdx.z;
  int rowBase = blockIdx.x * 32, colBase = ngroup * 64;
  const bf16_t* Az = A + (size_t)z * aSz;
  const bf16_t* Bz = Bw + (size_t)z * bSz;
  float* Cz = C + (size_t)z * cSz;
  v8f acc[8];
#pragma unroll
  for (int i = 0; i < 8; ++i) acc[i] = (v8f){0.f,0.f,0.f,0.f,0.f,0.f,0.f,0.f};
  wmma_2x4<LDA, LDB, KK>(Az, Bz, rowBase, colBase, acc);
  store_f32_2x4(Cz, ldc, rowBase, colBase, bias, nBias, nTilesValid, acc);
}

template <int LDA, int LDB, int KK>
__global__ void k_gemm_bf16out(const bf16_t* __restrict__ A, unsigned long long aSz,
                               const bf16_t* __restrict__ Bw, unsigned long long bSz,
                               bf16_t* __restrict__ C, int ldc, unsigned long long cSz,
                               int nTilesValid) {
  int wv = threadIdx.x >> 5, lane = threadIdx.x & 31;
  int ngroup = blockIdx.y * 4 + wv;
  if (ngroup * 4 >= nTilesValid) return;
  int z = blockIdx.z;
  int rowBase = blockIdx.x * 32, colBase = ngroup * 64;
  const bf16_t* Az = A + (size_t)z * aSz;
  const bf16_t* Bz = Bw + (size_t)z * bSz;
  bf16_t* Cz = C + (size_t)z * cSz;
  v8f acc[8];
#pragma unroll
  for (int i = 0; i < 8; ++i) acc[i] = (v8f){0.f,0.f,0.f,0.f,0.f,0.f,0.f,0.f};
  wmma_2x4<LDA, LDB, KK>(Az, Bz, rowBase, colBase, acc);
  const int half = lane >> 4, l15 = lane & 15;
#pragma unroll
  for (int mt = 0; mt < 2; ++mt)
#pragma unroll
    for (int nt = 0; nt < 4; ++nt) {
      int ntile = colBase / 16 + nt;
      if (ntile >= nTilesValid) continue;
      int n  = colBase + nt * 16 + l15;
      int m0 = rowBase + mt * 16 + half * 8;
      v8f a = acc[mt * 4 + nt];
#pragma unroll
      for (int r = 0; r < 8; ++r)
        Cz[(size_t)(m0 + r) * ldc + n] = (bf16_t)a[r];
    }
}

// ---------------- layer-0 per-step kernels ----------------
// attention: att[b,k] = sum_h tanh(proj[sidx,k,b,h] + ctx[d,b,h]) * ws[h] + bs
// softmax over k, then x[b,:] = sum_k w_k * Eg[sidx,k,b,:]
__global__ void k_att(const float* __restrict__ proj, const float* __restrict__ ctx,
                      const bf16_t* __restrict__ Eg, const float* __restrict__ wsv,
                      const float* __restrict__ bsv, const int* __restrict__ lengths,
                      bf16_t* __restrict__ xbf, int s) {
  int b = blockIdx.x, d = blockIdx.y, t = threadIdx.x;
  int L = lengths[b];
  int sidx = (d == 0) ? s : ((s < L) ? (L - 1 - s) : s);
  const float* ctxd = ctx + ((size_t)d * Bn + b) * Hp;
  __shared__ float red[256];
  __shared__ float wsh[5];
  float att[5];
  for (int k = 0; k < 5; ++k) {
    const float* p = proj + (((size_t)sidx * 5 + k) * Bn + b) * Hp;
    float partial = 0.0f;
    for (int h = t; h < Hn; h += 256)
      partial += tanhf(p[h] + ctxd[h]) * wsv[h];
    red[t] = partial;
    __syncthreads();
    for (int off = 128; off > 0; off >>= 1) {
      if (t < off) red[t] += red[t + off];
      __syncthreads();
    }
    att[k] = red[0] + bsv[0];
    __syncthreads();
  }
  if (t == 0) {
    float mx = att[0];
    for (int k = 1; k < 5; ++k) mx = fmaxf(mx, att[k]);
    float ssum = 0.0f, e[5];
    for (int k = 0; k < 5; ++k) { e[k] = __expf(att[k] - mx); ssum += e[k]; }
    for (int k = 0; k < 5; ++k) wsh[k] = e[k] / ssum;
  }
  __syncthreads();
  for (int e = t; e < Ep; e += 256) {
    float x = 0.0f;
    for (int k = 0; k < 5; ++k)
      x += wsh[k] * (float)Eg[(((size_t)sidx * 5 + k) * Bn + b) * Ep + e];
    xbf[((size_t)d * Bn + b) * Ep + e] = (bf16_t)x;   // Eg pad is 0 -> x pad is 0
  }
}

// gates = x @ Wih0^T + h @ Whh0^T + bias
__global__ void k_gates0(const bf16_t* __restrict__ xbf, const bf16_t* __restrict__ hbf,
                         const bf16_t* __restrict__ Wih, const bf16_t* __restrict__ Whh,
                         const float* __restrict__ bias, float* __restrict__ gates) {
  int wv = threadIdx.x >> 5;
  int ngroup = blockIdx.y * 4 + wv;
  if (ngroup * 4 >= Gn / 16) return;
  int d = blockIdx.z;
  int rowBase = blockIdx.x * 32, colBase = ngroup * 64;
  v8f acc[8];
#pragma unroll
  for (int i = 0; i < 8; ++i) acc[i] = (v8f){0.f,0.f,0.f,0.f,0.f,0.f,0.f,0.f};
  wmma_2x4<Ep, Ep, Ep>(xbf + (size_t)d * Bn * Ep,
                       Wih + (size_t)d * Gp * Ep, rowBase, colBase, acc);
  wmma_2x4<Hp, Hp, Hp>(hbf + (size_t)d * Bn * Hp,
                       Whh + (size_t)d * Gp * Hp, rowBase, colBase, acc);
  store_f32_2x4(gates + (size_t)d * Bn * Gp, Gp, rowBase, colBase,
                bias + (size_t)d * Gn, Gn, Gn / 16, acc);
}

__global__ void k_point0(const float* __restrict__ gates, float* __restrict__ h,
                         float* __restrict__ c, bf16_t* __restrict__ hbf,
                         bf16_t* __restrict__ cbf, bf16_t* __restrict__ x1,
                         const int* __restrict__ lengths, int s) {
  int d = blockIdx.z;
  int idx = blockIdx.x * blockDim.x + threadIdx.x;
  if (idx >= Bn * Hn) return;
  int b = idx / Hn, j = idx % Hn;
  int L = lengths[b];
  if (s >= L) return;                         // masked: freeze state, outputs stay 0
  const float* g = gates + ((size_t)d * Bn + b) * Gp;
  float gi = g[j], gf = g[Hn + j], gg = g[2 * Hn + j], go = g[3 * Hn + j];
  size_t off = ((size_t)d * Bn + b) * Hp + j;
  float c2 = sigm(gf) * c[off] + sigm(gi) * tanhf(gg);
  float h2 = sigm(go) * tanhf(c2);
  h[off] = h2; c[off] = c2;
  hbf[off] = (bf16_t)h2; cbf[off] = (bf16_t)c2;
  int pos = (d == 0) ? s : (L - 1 - s);       // bwd output lands time-reversed
  x1[((size_t)pos * Bn + b) * H2p + d * Hn + j] = (bf16_t)h2;
}

// ---------------- layer-1 per-step kernels ----------------
// gates = Gx[d][sidx(b)] + h1 @ Whh1^T + bias1
__global__ void k_gates1(const bf16_t* __restrict__ hbf1, const bf16_t* __restrict__ Whh,
                         const bf16_t* __restrict__ Gx, const float* __restrict__ bias,
                         const int* __restrict__ lengths, float* __restrict__ gates, int s) {
  int wv = threadIdx.x >> 5, lane = threadIdx.x & 31;
  int ngroup = blockIdx.y * 4 + wv;
  if (ngroup * 4 >= Gn / 16) return;
  int d = blockIdx.z;
  int rowBase = blockIdx.x * 32, colBase = ngroup * 64;
  v8f acc[8];
#pragma unroll
  for (int i = 0; i < 8; ++i) acc[i] = (v8f){0.f,0.f,0.f,0.f,0.f,0.f,0.f,0.f};
  wmma_2x4<Hp, Hp, Hp>(hbf1 + (size_t)d * Bn * Hp,
                       Whh + (size_t)d * Gp * Hp, rowBase, colBase, acc);
  const int half = lane >> 4, l15 = lane & 15;
  const bf16_t* Gxd = Gx + (size_t)d * Sn * Bn * Gp;
  float* Gz = gates + (size_t)d * Bn * Gp;
  const float* bz = bias + (size_t)d * Gn;
#pragma unroll
  for (int mt = 0; mt < 2; ++mt)
#pragma unroll
    for (int nt = 0; nt < 4; ++nt) {
      int ntile = colBase / 16 + nt;
      if (ntile >= Gn / 16) continue;
      int n  = colBase + nt * 16 + l15;
      int m0 = rowBase + mt * 16 + half * 8;
      float bv = bz[n];
      v8f a = acc[mt * 4 + nt];
#pragma unroll
      for (int r = 0; r < 8; ++r) {
        int m = m0 + r;                        // batch index
        int L = lengths[m];
        int sidx = (d == 0) ? s : ((s < L) ? (L - 1 - s) : s);
        float gx = (float)Gxd[((size_t)sidx * Bn + m) * Gp + n];
        Gz[(size_t)m * Gp + n] = a[r] + gx + bv;
      }
    }
}

__global__ void k_point1(const float* __restrict__ gates, float* __restrict__ h,
                         float* __restrict__ c, bf16_t* __restrict__ hbf,
                         const int* __restrict__ lengths, float* __restrict__ outp, int s) {
  int d = blockIdx.z;
  int idx = blockIdx.x * blockDim.x + threadIdx.x;
  if (idx >= Bn * Hn) return;
  int b = idx / Hn, j = idx % Hn;
  int L = lengths[b];
  if (s >= L) return;
  const float* g = gates + ((size_t)d * Bn + b) * Gp;
  float gi = g[j], gf = g[Hn + j], gg = g[2 * Hn + j], go = g[3 * Hn + j];
  size_t off = ((size_t)d * Bn + b) * Hp + j;
  float c2 = sigm(gf) * c[off] + sigm(gi) * tanhf(gg);
  float h2 = sigm(go) * tanhf(c2);
  h[off] = h2; c[off] = c2;
  hbf[off] = (bf16_t)h2;
  int pos = (d == 0) ? s : (L - 1 - s);
  outp[((size_t)pos * Bn + b) * (2 * Hn) + d * Hn + j] = h2;
}

// final_h = [h0f,h0b,h1f,h1b] (4,B,H); final_c analogous
__global__ void k_final(const float* __restrict__ h0, const float* __restrict__ c0,
                        const float* __restrict__ h1, const float* __restrict__ c1,
                        float* __restrict__ out) {
  int i = blockIdx.x * blockDim.x + threadIdx.x;
  if (i >= 4 * Bn * Hn) return;
  int j = i % Hn, b = (i / Hn) % Bn, r = i / (Bn * Hn);
  int d = r & 1;
  const float* hs = (r < 2) ? h0 : h1;
  const float* cs = (r < 2) ? c0 : c1;
  size_t off = ((size_t)d * Bn + b) * Hp + j;
  out[i] = hs[off];
  out[4 * Bn * Hn + i] = cs[off];
}

// ---------------- host launcher ----------------
extern "C" void kernel_launch(void* const* d_in, const int* in_sizes, int n_in,
                              void* d_out, int out_size, void* d_ws, size_t ws_size,
                              hipStream_t stream) {
  (void)in_sizes; (void)n_in; (void)ws_size;
  const int*   input   = (const int*)d_in[0];
  const int*   lengths = (const int*)d_in[2];
  const float* emb  = (const float*)d_in[3];
  const float* Wl1  = (const float*)d_in[4];
  const float* bl1  = (const float*)d_in[5];
  const float* Wc1  = (const float*)d_in[6];
  const float* bc1  = (const float*)d_in[7];
  const float* wsv  = (const float*)d_in[8];
  const float* bsv  = (const float*)d_in[9];
  const float* Wih0 = (const float*)d_in[10];
  const float* Whh0 = (const float*)d_in[11];
  const float* bih0 = (const float*)d_in[12];
  const float* bhh0 = (const float*)d_in[13];
  const float* Wih1 = (const float*)d_in[14];
  const float* Whh1 = (const float*)d_in[15];
  const float* bih1 = (const float*)d_in[16];
  const float* bhh1 = (const float*)d_in[17];
  float* out = (float*)d_out;

  // ---- workspace carve-up (256B aligned) ----
  char* p = (char*)d_ws;
  auto alloc = [&](size_t bytes) -> char* {
    char* r = p; p += (bytes + 255) & ~(size_t)255; return r;
  };
  bf16_t* wWl1T = (bf16_t*)alloc((size_t)Hp * Ep * 2);            // (512,512)
  bf16_t* wWc1T = (bf16_t*)alloc((size_t)Hp * Hp * 2);            // (512,512)
  bf16_t* wWih0 = (bf16_t*)alloc((size_t)2 * Gp * Ep * 2);        // (2,2048,512)
  bf16_t* wWhh0 = (bf16_t*)alloc((size_t)2 * Gp * Hp * 2);
  bf16_t* wWih1 = (bf16_t*)alloc((size_t)2 * Gp * H2p * 2);       // (2,2048,1024)
  bf16_t* wWhh1 = (bf16_t*)alloc((size_t)2 * Gp * Hp * 2);
  float*  bias0 = (float*)alloc((size_t)2 * Gn * 4);
  float*  bias1 = (float*)alloc((size_t)2 * Gn * 4);
  bf16_t* Eg    = (bf16_t*)alloc((size_t)Sn * 5 * Bn * Ep * 2);   // 41.9 MB
  float*  proj  = (float*)alloc((size_t)Sn * 5 * Bn * Hp * 4);    // 83.9 MB
  bf16_t* x1    = (bf16_t*)alloc((size_t)Sn * Bn * H2p * 2);      // 16.8 MB
  bf16_t* Gx    = (bf16_t*)alloc((size_t)2 * Sn * Bn * Gp * 2);   // 67.1 MB
  // contiguous zero-init state block
  char* stateBase = p;
  float*  h0   = (float*)alloc((size_t)2 * Bn * Hp * 4);
  float*  c0   = (float*)alloc((size_t)2 * Bn * Hp * 4);
  float*  h1   = (float*)alloc((size_t)2 * Bn * Hp * 4);
  float*  c1   = (float*)alloc((size_t)2 * Bn * Hp * 4);
  bf16_t* hbf0 = (bf16_t*)alloc((size_t)2 * Bn * Hp * 2);
  bf16_t* cbf0 = (bf16_t*)alloc((size_t)2 * Bn * Hp * 2);
  bf16_t* hbf1 = (bf16_t*)alloc((size_t)2 * Bn * Hp * 2);
  size_t stateBytes = (size_t)(p - stateBase);
  float*  ctx   = (float*)alloc((size_t)2 * Bn * Hp * 4);
  bf16_t* xbf   = (bf16_t*)alloc((size_t)2 * Bn * Ep * 2);
  float*  gates = (float*)alloc((size_t)2 * Bn * Gp * 4);

  // ---- zero init (every call: deterministic) ----
  k_zero_u32<<<1024, 256, 0, stream>>>((unsigned int*)out, (size_t)out_size);
  k_zero_u32<<<1024, 256, 0, stream>>>((unsigned int*)x1, (size_t)Sn * Bn * H2p / 2);
  k_zero_u32<<<256, 256, 0, stream>>>((unsigned int*)stateBase, stateBytes / 4);

  // ---- pack weights to bf16 (N,K) padded ----
  auto pg = [](size_t n) { return dim3((unsigned)((n + 255) / 256)); };
  k_pack<<<pg((size_t)Hp * Ep), 256, 0, stream>>>(Wl1, wWl1T, Hn, En, Hp, Ep, 1);
  k_pack<<<pg((size_t)Hp * Hp), 256, 0, stream>>>(Wc1, wWc1T, Hn, Hn, Hp, Hp, 1);
  for (int d = 0; d < 2; ++d) {
    k_pack<<<pg((size_t)Gp * Ep), 256, 0, stream>>>(Wih0 + (size_t)d * 4 * Hn * En,
        wWih0 + (size_t)d * Gp * Ep, Gn, En, Gp, Ep, 0);
    k_pack<<<pg((size_t)Gp * Hp), 256, 0, stream>>>(Whh0 + (size_t)d * 4 * Hn * Hn,
        wWhh0 + (size_t)d * Gp * Hp, Gn, Hn, Gp, Hp, 0);
    k_pack<<<pg((size_t)Gp * H2p), 256, 0, stream>>>(Wih1 + (size_t)d * 4 * Hn * 2 * Hn,
        wWih1 + (size_t)d * Gp * H2p, Gn, 2 * Hn, Gp, H2p, 0);
    k_pack<<<pg((size_t)Gp * Hp), 256, 0, stream>>>(Whh1 + (size_t)d * 4 * Hn * Hn,
        wWhh1 + (size_t)d * Gp * Hp, Gn, Hn, Gp, Hp, 0);
  }
  k_bias_add<<<pg(2 * Gn), 256, 0, stream>>>(bih0, bhh0, bias0, 2 * Gn);
  k_bias_add<<<pg(2 * Gn), 256, 0, stream>>>(bih1, bhh1, bias1, 2 * Gn);

  // ---- embedding gather (forward order; bwd indexes via ridx) ----
  k_gather<<<dim3(Sn * 5 * Bn), 128, 0, stream>>>(input, emb, Eg);

  // ---- proj = Eg @ Wl1 + bl1 : M=40960, N=512(500), K=512 ----
  k_gemm_f32<Ep, Ep, Ep><<<dim3(Sn * 5 * Bn / 32, 2, 1), 128, 0, stream>>>(
      Eg, 0ull, wWl1T, 0ull, bl1, Hn, proj, Hp, 0ull, Hp / 16);

  // ---- layer 0 recurrent scan (both dirs via blockIdx.z) ----
  for (int s = 0; s < Sn; ++s) {
    // ctx = c @ Wc1 + bc1
    k_gemm_f32<Hp, Hp, Hp><<<dim3(Bn / 32, 2, 2), 128, 0, stream>>>(
        cbf0, (unsigned long long)(Bn * Hp),
        wWc1T, 0ull, bc1, Hn,
        ctx, Hp, (unsigned long long)(Bn * Hp), Hp / 16);
    k_att<<<dim3(Bn, 2), 256, 0, stream>>>(proj, ctx, Eg, wsv, bsv, lengths, xbf, s);
    k_gates0<<<dim3(Bn / 32, 8, 2), 128, 0, stream>>>(xbf, hbf0, wWih0, wWhh0, bias0, gates);
    k_point0<<<dim3(Bn * Hn / 256, 1, 2), 256, 0, stream>>>(gates, h0, c0, hbf0, cbf0, x1,
                                                            lengths, s);
  }

  // ---- hoisted layer-1 input GEMM: Gx[d] = x1 @ Wih1[d]^T (M=8192,K=1024,N=2000) ----
  k_gemm_bf16out<H2p, H2p, H2p><<<dim3(Sn * Bn / 32, 8, 2), 128, 0, stream>>>(
      x1, 0ull, wWih1, (unsigned long long)((size_t)Gp * H2p),
      Gx, Gp, (unsigned long long)((size_t)Sn * Bn * Gp), Gn / 16);

  // ---- layer 1 recurrent scan ----
  float* outputs = out + (size_t)2 * 4 * Bn * Hn;   // after final_h, final_c
  for (int s = 0; s < Sn; ++s) {
    k_gates1<<<dim3(Bn / 32, 8, 2), 128, 0, stream>>>(hbf1, wWhh1, Gx, bias1,
                                                      lengths, gates, s);
    k_point1<<<dim3(Bn * Hn / 256, 1, 2), 256, 0, stream>>>(gates, h1, c1, hbf1,
                                                            lengths, outputs, s);
  }

  // ---- finals ----
  k_final<<<dim3(4 * Bn * Hn / 256), 256, 0, stream>>>(h0, c0, h1, c1, out);
}
